// TransformerLM_71330816852066
// MI455X (gfx1250) — matrix operand order repaired and checked by reading, not compiled, not run on hardware
//
#include <hip/hip_runtime.h>
#include <hip/hip_bf16.h>
#include <math.h>

// ---------------------------------------------------------------------------
// Problem constants (from the reference)
// ---------------------------------------------------------------------------
#define L_LAYERS 6
#define D_MODEL  1024
#define NHEADS   16
#define FFN_F    4096
#define VOCAB    32000
#define SEQ      1024
#define BATCH    2
#define HDIM     64
#define ROWS     (BATCH * SEQ)    // 2048 token rows
#define BHZ      (BATCH * NHEADS) // 32 attention batches

// ---------------------------------------------------------------------------
// WMMA types (CDNA5 / gfx1250, wave32)
// ---------------------------------------------------------------------------
typedef __attribute__((ext_vector_type(4)))  __bf16 v4bf;
typedef __attribute__((ext_vector_type(8)))  __bf16 v8bf;
typedef __attribute__((ext_vector_type(16))) __bf16 v16bf;
typedef __attribute__((ext_vector_type(4)))  float  v4f;
typedef __attribute__((ext_vector_type(8)))  float  v8f;

constexpr int BM = 128;
constexpr int BN = 128;
constexpr int BK = 32;
constexpr int LST = 40;  // LDS row stride (bf16): 80B rows -> every access 16B aligned

enum : int { EPI_PLAIN = 0, EPI_SCORES = 1, EPI_GELU = 2, EPI_GATE_RES = 3, EPI_BF16 = 4 };

__device__ __forceinline__ float gelu_exact(float x) {
  return 0.5f * x * (1.0f + erff(x * 0.70710678118654752f));
}

// ---------------------------------------------------------------------------
// Tiled WMMA GEMM, fully compile-time layout.
//   C[z] = epilogue(A[z] @ B[z] + bias)
//   A: bf16 [M x K] row-major (lda = LDA), per-z stride SA.
//   B: if BTR: bf16 [N x K] row-major (k^T case); else [K x N] row-major,
//      fp32 (weights) or bf16 (BF16B), converted to bf16 while staging to LDS.
//   256 threads = 8 waves (wave32); wave grid 4(m) x 2(n); each wave owns a
//   32x64 patch = 2x4 tiles of v_wmma_f32_16x16x32_bf16.
// ---------------------------------------------------------------------------
template <int M, int N, int K, int LDA, int LDB, int LDC,
          bool BF16B, bool BTR, int EPI, int ZDIV,
          long long SA, long long SB, long long SCO, long long SCI>
__global__ __launch_bounds__(256) void wmma_gemm(
    const __bf16* __restrict__ A, const void* __restrict__ Bv,
    const float* __restrict__ bias, float* __restrict__ Cf,
    __bf16* __restrict__ Cbf, const float* __restrict__ resid,
    const float* __restrict__ gate) {
  constexpr bool NGUARD = (N % BN) != 0;
  __shared__ __bf16 As[BM * LST];
  __shared__ __bf16 Bs[BN * LST];

  const int z  = blockIdx.z;
  const int m0 = blockIdx.y * BM;
  const int n0 = blockIdx.x * BN;
  if constexpr (EPI == EPI_SCORES) {
    if (n0 > m0 + BM - 1) return;  // tile fully above the causal diagonal
  }

  const int tid  = threadIdx.x;
  const int lane = tid & 31, wid = tid >> 5;
  const int wm = wid & 3, wn = wid >> 2;
  const int rsel = lane & 15, hi = lane >> 4;

  // ---- staging coordinates (hoisted out of the k-loop) ----
  const int ar = tid >> 1;         // 0..127 : A row (or B column when BTR)
  const int ah = (tid & 1) * 16;   // 0 / 16 : k-offset within the BK tile
  const __bf16* ap = A + (long long)z * SA + (long long)(m0 + ar) * LDA + ah;
  __bf16* as_dst = &As[ar * LST + ah];

  const __bf16* bp_dir = nullptr;  // BTR direct path
  __bf16* bs_dst = &Bs[ar * LST + ah];
  if constexpr (BTR)
    bp_dir = (const __bf16*)Bv + (long long)z * SB + (long long)(n0 + ar) * LDB + ah;

  const int n4 = (tid & 31) * 4;   // scatter path: 4 consecutive n per thread
  const int kr = tid >> 5;         // scatter path: k = kr + 8*i

  // ---- fragment LDS pointers (ISA 7.12.2 16-bit layouts) ----
  const __bf16* afp0 = &As[(wm * 32 + rsel) * LST + hi * 8];
  const __bf16* afp1 = afp0 + 16 * LST;
  const __bf16* bfp  = &Bs[(wn * 64 + rsel) * LST + hi * 8];

  v8f acc[2][4];
#pragma unroll
  for (int mi = 0; mi < 2; ++mi)
#pragma unroll
    for (int ni = 0; ni < 4; ++ni)
#pragma unroll
      for (int e = 0; e < 8; ++e) acc[mi][ni][e] = 0.0f;

  for (int k0 = 0; k0 < K; k0 += BK) {
    if (k0 + BK < K)  // global_prefetch_b8 of next A tile
      __builtin_prefetch((const void*)(ap + BK), 0, 1);

    // Stage A: two 16B loads + two 16B LDS stores per thread.
    {
      v8bf a0 = *(const v8bf*)ap;
      v8bf a1 = *(const v8bf*)(ap + 8);
      *(v8bf*)as_dst = a0;
      *(v8bf*)(as_dst + 8) = a1;
      ap += BK;
    }
    // Stage B.
    if constexpr (BTR) {  // bf16 [N x K]: same vector path as A
      v8bf b0 = *(const v8bf*)bp_dir;
      v8bf b1 = *(const v8bf*)(bp_dir + 8);
      *(v8bf*)bs_dst = b0;
      *(v8bf*)(bs_dst + 8) = b1;
      bp_dir += BK;
    } else if constexpr (BF16B) {  // bf16 [K x N]: vector-read n, scatter k
      const __bf16* bg = (const __bf16*)Bv + (long long)z * SB;
#pragma unroll
      for (int i = 0; i < 4; ++i) {
        const int kk = kr + 8 * i;
        v4bf v;
#pragma unroll
        for (int j = 0; j < 4; ++j) v[j] = (__bf16)0.0f;
        bool ok = true;
        if constexpr (NGUARD) ok = (n0 + n4 + 3) < N;
        if (ok) v = *(const v4bf*)(bg + (long long)(k0 + kk) * LDB + (n0 + n4));
#pragma unroll
        for (int j = 0; j < 4; ++j) Bs[(n4 + j) * LST + kk] = v[j];
      }
    } else {  // fp32 weights [K x N]: float4 read, bf16-convert, scatter
      const float* bg = (const float*)Bv;
#pragma unroll
      for (int i = 0; i < 4; ++i) {
        const int kk = kr + 8 * i;
        v4f v = {0.0f, 0.0f, 0.0f, 0.0f};
        bool ok = true;
        if constexpr (NGUARD) ok = (n0 + n4 + 3) < N;
        if (ok) v = *(const v4f*)(bg + (long long)(k0 + kk) * LDB + (n0 + n4));
#pragma unroll
        for (int j = 0; j < 4; ++j) Bs[(n4 + j) * LST + kk] = (__bf16)v[j];
      }
    }
    __syncthreads();

    // A fragments (kept live); B fragments loaded one-at-a-time.
    v8bf alo0 = *(const v8bf*)afp0, ahi0 = *(const v8bf*)(afp0 + 16);
    v8bf alo1 = *(const v8bf*)afp1, ahi1 = *(const v8bf*)(afp1 + 16);
    v16bf a0 = __builtin_shufflevector(alo0, ahi0, 0, 1, 2, 3, 4, 5, 6, 7,
                                       8, 9, 10, 11, 12, 13, 14, 15);
    v16bf a1 = __builtin_shufflevector(alo1, ahi1, 0, 1, 2, 3, 4, 5, 6, 7,
                                       8, 9, 10, 11, 12, 13, 14, 15);
#pragma unroll
    for (int ni = 0; ni < 4; ++ni) {
      v8bf blo = *(const v8bf*)(bfp + ni * 16 * LST);
      v8bf bhi = *(const v8bf*)(bfp + ni * 16 * LST + 16);
      v16bf b = __builtin_shufflevector(blo, bhi, 0, 1, 2, 3, 4, 5, 6, 7,
                                        8, 9, 10, 11, 12, 13, 14, 15);
      acc[0][ni] = __builtin_amdgcn_wmma_f32_16x16x32_bf16(
          false, a0, false, b, (short)0, acc[0][ni], false, false);
      acc[1][ni] = __builtin_amdgcn_wmma_f32_16x16x32_bf16(
          false, a1, false, b, (short)0, acc[1][ni], false, false);
    }
    __syncthreads();
  }

  // ---- epilogue (C layout: VGPR e -> row e, +8 for upper lane half) ----
  const long long cbase =
      (long long)(z / ZDIV) * SCO + (long long)(z % ZDIV) * SCI;
#pragma unroll
  for (int mi = 0; mi < 2; ++mi)
#pragma unroll
    for (int ni = 0; ni < 4; ++ni) {
      const int nc = n0 + wn * 64 + ni * 16 + rsel;
      if constexpr (NGUARD) {
        if (nc >= N) continue;
      }
      const int mb = m0 + wm * 32 + mi * 16 + hi * 8;
      float bval = 0.0f;
      if constexpr (EPI != EPI_SCORES)
        if (bias != nullptr) bval = bias[nc];
#pragma unroll
      for (int e = 0; e < 8; ++e) {
        const int m = mb + e;
        const float v = acc[mi][ni][e];
        const long long ci = cbase + (long long)m * LDC + nc;
        if constexpr (EPI == EPI_PLAIN) {
          Cf[ci] = v + bval;
        } else if constexpr (EPI == EPI_SCORES) {
          Cf[ci] = v * 0.125f;  // 1/sqrt(HDIM=64)
        } else if constexpr (EPI == EPI_GELU) {
          Cbf[ci] = (__bf16)gelu_exact(v + bval);
        } else if constexpr (EPI == EPI_BF16) {
          Cbf[ci] = (__bf16)(v + bval);
        } else {  // EPI_GATE_RES
          const int b = m / SEQ;
          Cf[ci] = resid[(long long)m * LDC + nc] + (v + bval) * gate[b * N + nc];
        }
      }
    }
}

// ---------------------------------------------------------------------------
// Embedding gather: x[row] = tok_emb[ids[row]]
// ---------------------------------------------------------------------------
__global__ __launch_bounds__(256) void embed_kernel(const int* ids,
                                                    const float* emb,
                                                    float* x) {
  const long long row = blockIdx.x;
  const long long id = ids[row];
  const float* e = emb + id * D_MODEL;
  float* xr = x + row * D_MODEL;
  for (int c = threadIdx.x; c < D_MODEL; c += 256) xr[c] = e[c];
}

// ---------------------------------------------------------------------------
// LayerNorm (fp32 in -> bf16 out), one 256-thread block per row.
// ---------------------------------------------------------------------------
__global__ __launch_bounds__(256) void layernorm_kernel(const float* x,
                                                        const float* g,
                                                        const float* b,
                                                        __bf16* out) {
  __shared__ float red[256];
  const long long row = blockIdx.x;
  const float* xr = x + row * D_MODEL;
  const int tid = threadIdx.x;
  float s = 0.0f, s2 = 0.0f;
  for (int c = tid; c < D_MODEL; c += 256) {
    const float v = xr[c];
    s += v; s2 += v * v;
  }
  red[tid] = s; __syncthreads();
  for (int w = 128; w > 0; w >>= 1) { if (tid < w) red[tid] += red[tid + w]; __syncthreads(); }
  s = red[0]; __syncthreads();
  red[tid] = s2; __syncthreads();
  for (int w = 128; w > 0; w >>= 1) { if (tid < w) red[tid] += red[tid + w]; __syncthreads(); }
  s2 = red[0];
  const float mean = s * (1.0f / D_MODEL);
  const float var  = s2 * (1.0f / D_MODEL) - mean * mean;
  const float rst  = rsqrtf(var + 1e-5f);
  for (int c = tid; c < D_MODEL; c += 256)
    out[row * D_MODEL + c] = (__bf16)((xr[c] - mean) * rst * g[c] + b[c]);
}

// ---------------------------------------------------------------------------
// RoPE on q,k + head-major bf16 repack of q,k,v:  [B,S,H,HD] -> [B,H,S,HD]
// ---------------------------------------------------------------------------
__global__ __launch_bounds__(256) void rope_pack_kernel(
    const float* qf, const float* kf, const float* vf,
    const float* cosT, const float* sinT,
    __bf16* qbf, __bf16* kbf, __bf16* vbf) {
  const long long row = blockIdx.x;          // b*S + s
  const int b = (int)(row / SEQ), s = (int)(row % SEQ);
  const long long base = row * D_MODEL;
  for (int col = threadIdx.x; col < D_MODEL; col += 256) {
    const int h = col >> 6, j = col & 63;
    const float cv = cosT[s * HDIM + j];
    const float sv = sinT[s * HDIM + j];
    const int jp = (j < 32) ? (col + 32) : (col - 32);
    const float qv = qf[base + col];
    const float kv = kf[base + col];
    const float qr = (j < 32) ? -qf[base + jp] : qf[base + jp];
    const float kr = (j < 32) ? -kf[base + jp] : kf[base + jp];
    const long long dst = (((long long)(b * NHEADS + h)) * SEQ + s) * HDIM + j;
    qbf[dst] = (__bf16)(qv * cv + qr * sv);
    kbf[dst] = (__bf16)(kv * cv + kr * sv);
    vbf[dst] = (__bf16)vf[base + col];
  }
}

// ---------------------------------------------------------------------------
// Causal softmax over one score row -> bf16 probs (zeros past the diagonal).
// ---------------------------------------------------------------------------
__global__ __launch_bounds__(256) void softmax_kernel(const float* scores,
                                                      __bf16* probs) {
  __shared__ float red[256];
  const long long row = blockIdx.x;          // z*S + r
  const int r = (int)(row % SEQ);
  const int valid = r + 1;
  const float* srow = scores + row * SEQ;
  __bf16* prow = probs + row * SEQ;
  const int tid = threadIdx.x;

  float m = -INFINITY;
  for (int c = tid; c < valid; c += 256) m = fmaxf(m, srow[c]);
  red[tid] = m; __syncthreads();
  for (int w = 128; w > 0; w >>= 1) { if (tid < w) red[tid] = fmaxf(red[tid], red[tid + w]); __syncthreads(); }
  m = red[0]; __syncthreads();

  float sum = 0.0f;
  for (int c = tid; c < valid; c += 256) sum += expf(srow[c] - m);
  red[tid] = sum; __syncthreads();
  for (int w = 128; w > 0; w >>= 1) { if (tid < w) red[tid] += red[tid + w]; __syncthreads(); }
  const float inv = 1.0f / red[0];

  for (int c = tid; c < SEQ; c += 256)
    prow[c] = (c < valid) ? (__bf16)(expf(srow[c] - m) * inv) : (__bf16)0.0f;
}

// ---------------------------------------------------------------------------
// Per-layer gate MLPs: sigmoid(gelu(sig@W1+b1)@W2+b2) for attn & ffn gates.
// One block per (layer, batch).
// ---------------------------------------------------------------------------
__global__ __launch_bounds__(256) void gates_kernel(
    const float* sig,
    const float* agW1, const float* agb1, const float* agW2, const float* agb2,
    const float* fgW1, const float* fgb1, const float* fgW2, const float* fgb2,
    float* ag_out, float* fg_out) {
  __shared__ float hA[32], hF[32];
  const int l = blockIdx.x / BATCH, b = blockIdx.x % BATCH;
  const float s0 = sig[b * 3 + 0], s1 = sig[b * 3 + 1], s2 = sig[b * 3 + 2];
  const int t = threadIdx.x;
  if (t < 32) {
    const float* W = agW1 + (long long)l * 3 * 32;
    hA[t] = gelu_exact(s0 * W[t] + s1 * W[32 + t] + s2 * W[64 + t] +
                       agb1[l * 32 + t]);
  } else if (t < 64) {
    const int j = t - 32;
    const float* W = fgW1 + (long long)l * 3 * 32;
    hF[j] = gelu_exact(s0 * W[j] + s1 * W[32 + j] + s2 * W[64 + j] +
                       fgb1[l * 32 + j]);
  }
  __syncthreads();
  for (int n = t; n < D_MODEL; n += 256) {
    float aa = agb2[(long long)l * D_MODEL + n];
    float ff = fgb2[(long long)l * D_MODEL + n];
    for (int j = 0; j < 32; ++j) {
      aa += hA[j] * agW2[((long long)l * 32 + j) * D_MODEL + n];
      ff += hF[j] * fgW2[((long long)l * 32 + j) * D_MODEL + n];
    }
    ag_out[((long long)l * BATCH + b) * D_MODEL + n] = 1.0f / (1.0f + expf(-aa));
    fg_out[((long long)l * BATCH + b) * D_MODEL + n] = 1.0f / (1.0f + expf(-ff));
  }
}

// ---------------------------------------------------------------------------
// Host orchestration
// ---------------------------------------------------------------------------
extern "C" void kernel_launch(void* const* d_in, const int* in_sizes, int n_in,
                              void* d_out, int out_size, void* d_ws,
                              size_t ws_size, hipStream_t stream) {
  (void)in_sizes; (void)n_in; (void)out_size; (void)ws_size;
  const int*   input_ids = (const int*)d_in[0];
  const float* gate_sig  = (const float*)d_in[1];
  const float* tok_emb   = (const float*)d_in[2];
  const float* rope_cos  = (const float*)d_in[3];
  const float* rope_sin  = (const float*)d_in[4];
  const float* Wq = (const float*)d_in[5];   const float* bq = (const float*)d_in[6];
  const float* Wk = (const float*)d_in[7];   const float* bk = (const float*)d_in[8];
  const float* Wv = (const float*)d_in[9];   const float* bv = (const float*)d_in[10];
  const float* Wo = (const float*)d_in[11];  const float* bo = (const float*)d_in[12];
  const float* ln1_g = (const float*)d_in[13]; const float* ln1_b = (const float*)d_in[14];
  const float* ln2_g = (const float*)d_in[15]; const float* ln2_b = (const float*)d_in[16];
  const float* W1 = (const float*)d_in[17];  const float* b1 = (const float*)d_in[18];
  const float* W2 = (const float*)d_in[19];  const float* b2 = (const float*)d_in[20];
  const float* agW1 = (const float*)d_in[21]; const float* agb1 = (const float*)d_in[22];
  const float* agW2 = (const float*)d_in[23]; const float* agb2 = (const float*)d_in[24];
  const float* fgW1 = (const float*)d_in[25]; const float* fgb1 = (const float*)d_in[26];
  const float* fgW2 = (const float*)d_in[27]; const float* fgb2 = (const float*)d_in[28];
  const float* lnf_g = (const float*)d_in[29]; const float* lnf_b = (const float*)d_in[30];
  const float* Wout = (const float*)d_in[31]; const float* bout = (const float*)d_in[32];

  // Workspace carving (256B aligned).
  size_t off = 0;
  auto carve = [&](size_t bytes) -> char* {
    char* p = (char*)d_ws + off;
    off += (bytes + 255) & ~(size_t)255;
    return p;
  };
  float*  xa     = (float*)carve((size_t)ROWS * D_MODEL * 4);
  float*  xb     = (float*)carve((size_t)ROWS * D_MODEL * 4);
  __bf16* nbf    = (__bf16*)carve((size_t)ROWS * D_MODEL * 2);
  float*  qf     = (float*)carve((size_t)ROWS * D_MODEL * 4);
  float*  kf     = (float*)carve((size_t)ROWS * D_MODEL * 4);
  float*  vf     = (float*)carve((size_t)ROWS * D_MODEL * 4);
  __bf16* qbf    = (__bf16*)carve((size_t)ROWS * D_MODEL * 2);
  __bf16* kbf    = (__bf16*)carve((size_t)ROWS * D_MODEL * 2);
  __bf16* vbf    = (__bf16*)carve((size_t)ROWS * D_MODEL * 2);
  float*  scores = (float*)carve((size_t)BHZ * SEQ * SEQ * 4);   // 134 MB (L2-resident)
  __bf16* probs  = (__bf16*)carve((size_t)BHZ * SEQ * SEQ * 2);  // 67 MB
  __bf16* aobf   = (__bf16*)carve((size_t)ROWS * D_MODEL * 2);
  __bf16* hbf    = (__bf16*)carve((size_t)ROWS * FFN_F * 2);
  float*  agv    = (float*)carve((size_t)L_LAYERS * BATCH * D_MODEL * 4);
  float*  fgv    = (float*)carve((size_t)L_LAYERS * BATCH * D_MODEL * 4);

  const dim3 blk(256);

  embed_kernel<<<ROWS, blk, 0, stream>>>(input_ids, tok_emb, xa);
  gates_kernel<<<L_LAYERS * BATCH, blk, 0, stream>>>(
      gate_sig, agW1, agb1, agW2, agb2, fgW1, fgb1, fgW2, fgb2, agv, fgv);

  for (int i = 0; i < L_LAYERS; ++i) {
    const float* Wq_i = Wq + (size_t)i * D_MODEL * D_MODEL;
    const float* Wk_i = Wk + (size_t)i * D_MODEL * D_MODEL;
    const float* Wv_i = Wv + (size_t)i * D_MODEL * D_MODEL;
    const float* Wo_i = Wo + (size_t)i * D_MODEL * D_MODEL;
    const float* W1_i = W1 + (size_t)i * D_MODEL * FFN_F;
    const float* W2_i = W2 + (size_t)i * FFN_F * D_MODEL;
    const float* agv_i = agv + (size_t)i * BATCH * D_MODEL;
    const float* fgv_i = fgv + (size_t)i * BATCH * D_MODEL;

    // ---- LN1 -> bf16 activations
    layernorm_kernel<<<ROWS, blk, 0, stream>>>(
        xa, ln1_g + (size_t)i * D_MODEL, ln1_b + (size_t)i * D_MODEL, nbf);

    // ---- Q/K/V projections
    {
      dim3 grid(D_MODEL / BN, ROWS / BM, 1);
      wmma_gemm<ROWS, D_MODEL, D_MODEL, D_MODEL, D_MODEL, D_MODEL,
                false, false, EPI_PLAIN, 1, 0, 0, 0, 0>
          <<<grid, blk, 0, stream>>>(nbf, Wq_i, bq + (size_t)i * D_MODEL, qf,
                                     nullptr, nullptr, nullptr);
      wmma_gemm<ROWS, D_MODEL, D_MODEL, D_MODEL, D_MODEL, D_MODEL,
                false, false, EPI_PLAIN, 1, 0, 0, 0, 0>
          <<<grid, blk, 0, stream>>>(nbf, Wk_i, bk + (size_t)i * D_MODEL, kf,
                                     nullptr, nullptr, nullptr);
      wmma_gemm<ROWS, D_MODEL, D_MODEL, D_MODEL, D_MODEL, D_MODEL,
                false, false, EPI_PLAIN, 1, 0, 0, 0, 0>
          <<<grid, blk, 0, stream>>>(nbf, Wv_i, bv + (size_t)i * D_MODEL, vf,
                                     nullptr, nullptr, nullptr);
    }

    // ---- RoPE + head-major bf16 pack
    rope_pack_kernel<<<ROWS, blk, 0, stream>>>(qf, kf, vf, rope_cos, rope_sin,
                                               qbf, kbf, vbf);

    // ---- scores = 0.125 * q @ k^T  (batched over B*H, causal tiles skipped)
    wmma_gemm<SEQ, SEQ, HDIM, HDIM, HDIM, SEQ, true, true, EPI_SCORES, 1,
              (long long)SEQ * HDIM, (long long)SEQ * HDIM,
              (long long)SEQ * SEQ, 0>
        <<<dim3(SEQ / BN, SEQ / BM, BHZ), blk, 0, stream>>>(
            qbf, kbf, nullptr, scores, nullptr, nullptr, nullptr);

    // ---- causal softmax -> bf16 probs
    softmax_kernel<<<BHZ * SEQ, blk, 0, stream>>>(scores, probs);

    // ---- ao = probs @ v  (bf16 out, written directly into [B,S,D] layout)
    wmma_gemm<SEQ, HDIM, SEQ, SEQ, HDIM, D_MODEL, true /*BF16B*/, false,
              EPI_BF16, NHEADS, (long long)SEQ * SEQ, (long long)SEQ * HDIM,
              (long long)SEQ * D_MODEL, (long long)HDIM>
        <<<dim3(1, SEQ / BM, BHZ), blk, 0, stream>>>(
            probs, vbf, nullptr, nullptr, aobf, nullptr, nullptr);

    // ---- xb = xa + (ao @ Wo + bo) * attn_gate
    wmma_gemm<ROWS, D_MODEL, D_MODEL, D_MODEL, D_MODEL, D_MODEL,
              false, false, EPI_GATE_RES, 1, 0, 0, 0, 0>
        <<<dim3(D_MODEL / BN, ROWS / BM, 1), blk, 0, stream>>>(
            aobf, Wo_i, bo + (size_t)i * D_MODEL, xb, nullptr, xa, agv_i);

    // ---- LN2
    layernorm_kernel<<<ROWS, blk, 0, stream>>>(
        xb, ln2_g + (size_t)i * D_MODEL, ln2_b + (size_t)i * D_MODEL, nbf);

    // ---- FFN1: h = gelu(n @ W1 + b1) -> bf16
    wmma_gemm<ROWS, FFN_F, D_MODEL, D_MODEL, FFN_F, FFN_F,
              false, false, EPI_GELU, 1, 0, 0, 0, 0>
        <<<dim3(FFN_F / BN, ROWS / BM, 1), blk, 0, stream>>>(
            nbf, W1_i, b1 + (size_t)i * FFN_F, nullptr, hbf, nullptr, nullptr);

    // ---- FFN2: xa = xb + (h @ W2 + b2) * ffn_gate
    wmma_gemm<ROWS, D_MODEL, FFN_F, FFN_F, D_MODEL, D_MODEL,
              false, false, EPI_GATE_RES, 1, 0, 0, 0, 0>
        <<<dim3(D_MODEL / BN, ROWS / BM, 1), blk, 0, stream>>>(
            hbf, W2_i, b2 + (size_t)i * D_MODEL, xa, nullptr, xb, fgv_i);
  }

  // ---- Final LN + LM head: logits = lnf(x) @ Wout + bout
  layernorm_kernel<<<ROWS, blk, 0, stream>>>(xa, lnf_g, lnf_b, nbf);
  wmma_gemm<ROWS, VOCAB, D_MODEL, D_MODEL, VOCAB, VOCAB,
            false, false, EPI_PLAIN, 1, 0, 0, 0, 0>
      <<<dim3(VOCAB / BN, ROWS / BM, 1), blk, 0, stream>>>(
          nbf, Wout, bout, (float*)d_out, nullptr, nullptr, nullptr);
}
